// VectorizedExpertMLP_35725537968798
// MI455X (gfx1250) — matrix operand order repaired and checked by reading, not compiled, not run on hardware
//
#include <hip/hip_runtime.h>
#include <hip/hip_bf16.h>

// MoE dense-expert MLP for MI455X (gfx1250, wave32, WMMA + async LDS DMA).
// E=8, D=1024, H=4096, N=B*S=4096. All dims divisible by tile sizes.
//
// Block: 128 threads = 4 waves (2x2), block tile 128x128, K-step 32.
// Wave tile 64x64 -> 16 v_wmma per K-step per wave from 8 fragment loads.
// A-tile staged with GLOBAL_LOAD_ASYNC_TO_LDS_B128 (ASYNCcnt); B-tile staged
// through VGPRs (f32 -> bf16 convert + transpose). LDS double-buffered ~39 KB.

typedef __attribute__((ext_vector_type(16))) __bf16 v16bf;
typedef __attribute__((ext_vector_type(8)))  float  v8f;
typedef __attribute__((ext_vector_type(4)))  int    v4i;
typedef unsigned int u32;

#define TBM 128    // block tile M (tokens)
#define TBN 128    // block tile N
#define TBK 32     // K per WMMA step (16x16x32 bf16)
#define APITCH 40  // A LDS row pitch in bf16 (32+8 pad): 80B rows, 16B aligned
                   // (required for async B128 LDS writes; enables ds_load_b128)
#define BPITCH 36  // B LDS row pitch in bf16 (32+4 pad): 72B rows, 8B aligned;
                   // dword stride 18 -> conflict-free wave32 LDS stores

#if defined(__gfx1250__) && __has_builtin(__builtin_amdgcn_global_load_async_to_lds_b128)
#define ASYNC_A 1
#else
#define ASYNC_A 0
#endif

__device__ __forceinline__ void wait_async0() {
#if __has_builtin(__builtin_amdgcn_s_wait_asynccnt)
    __builtin_amdgcn_s_wait_asynccnt(0);
#else
    asm volatile("s_wait_asynccnt 0x0" ::: "memory");
#endif
}

__device__ __forceinline__ u32 pack2_bf16(float a, float b) {
    unsigned short ua = __builtin_bit_cast(unsigned short, (__bf16)a);
    unsigned short ub = __builtin_bit_cast(unsigned short, (__bf16)b);
    return ((u32)ub << 16) | (u32)ua;
}

// EPI==0: hidden = silu(acc) -> bf16 store (GEMM1)
// EPI==1: out (f32) = [accum ? += : =] rw[row,e] * acc (GEMM2)
template <int EPI>
__global__ __launch_bounds__(128) void moe_wmma_gemm(
    const __bf16* __restrict__ A,   // bf16, row-major, row stride = 2*lda2 elems
    const float*  __restrict__ W,   // f32 weights, row-major [K x ldb]
    void*         __restrict__ outPtr,
    const float*  __restrict__ rw,  // [Ntok x 8] routing weights (EPI==1)
    int lda2,                        // A row stride in dwords (elems/2)
    int ldb,                         // W row stride in elems
    int nKB,                         // number of K blocks (K / 32)
    int outLd,                       // output row stride in elems
    int expert, int accum)
{
    __shared__ u32 sA[2][TBM * (APITCH / 2)];   // 2 x 10240 B, rows 16B-aligned
    __shared__ u32 sB[2][TBN * (BPITCH / 2)];   // 2 x  9216 B

    const int tid  = threadIdx.x;
    const int lane = tid & 31;
    const int wave = tid >> 5;           // 0..3
    const int wr   = wave >> 1;          // 0..1 : 64-row band
    const int wc   = wave & 1;           // 0..1 : 64-col band
    const int half = lane >> 4;          // lane 0-15 vs 16-31
    const int l16  = lane & 15;

    const int rowBase = blockIdx.y * TBM;
    const int colBase = blockIdx.x * TBN;

    const int b_n = tid;                 // 0..127 : N within B tile
    u32 bR[16];                          // B staging registers

#if ASYNC_A
    // A tile 128x32 bf16 = 8 KB; 128 threads x 4 issues x 16B (B128) per K-step.
    const int a_c  = tid & 3;            // 16B chunk within row (0..3)
    const int a_r0 = tid >> 2;           // 0..31, rows r0 + 32*j
    auto gloadA_async = [&](int kb, int buf) {
        #pragma unroll
        for (int j = 0; j < 4; ++j) {
            const int r = a_r0 + 32 * j;
            const u32* g = (const u32*)A + (size_t)(rowBase + r) * (size_t)lda2
                           + (size_t)kb * 16 + a_c * 4;
            u32* l = &sA[buf][r * (APITCH / 2) + a_c * 4];
            __builtin_amdgcn_global_load_async_to_lds_b128(
                (__attribute__((address_space(1))) v4i*)g,
                (__attribute__((address_space(3))) v4i*)l,
                0, 0);
        }
    };
#else
    u32 aR[16];
    const int a_kp = tid & 15;           // dword column within K-block
    const int a_m0 = tid >> 4;           // 0..7, rows m0 + 8*i
    auto gloadA = [&](int kb) {
        const u32* Ag = (const u32*)A;
        size_t base = (size_t)rowBase * (size_t)lda2 + (size_t)kb * 16 + a_kp;
        #pragma unroll
        for (int i = 0; i < 16; ++i)
            aR[i] = Ag[base + (size_t)(a_m0 + 8 * i) * (size_t)lda2];
    };
    auto stashA = [&](int buf) {
        #pragma unroll
        for (int i = 0; i < 16; ++i)
            sA[buf][(a_m0 + 8 * i) * (APITCH / 2) + a_kp] = aR[i];
    };
#endif

    auto gloadB = [&](int kb) {
        const float* Wp = W + (size_t)(kb * TBK) * (size_t)ldb + colBase + b_n;
        #pragma unroll
        for (int i = 0; i < 16; ++i) {
            float f0 = Wp[(size_t)(2 * i) * (size_t)ldb];
            float f1 = Wp[(size_t)(2 * i + 1) * (size_t)ldb];
            bR[i] = pack2_bf16(f0, f1);   // LDS layout: [N][K] (transpose here)
        }
    };
    auto stashB = [&](int buf) {
        #pragma unroll
        for (int i = 0; i < 16; ++i)
            sB[buf][b_n * (BPITCH / 2) + i] = bR[i];
    };

    v8f acc[4][4];
    #pragma unroll
    for (int mt = 0; mt < 4; ++mt)
        #pragma unroll
        for (int nt = 0; nt < 4; ++nt)
            #pragma unroll
            for (int j = 0; j < 8; ++j)
                acc[mt][nt][j] = 0.0f;

    // ---- pipeline prologue ----
#if ASYNC_A
    gloadA_async(0, 0);
    gloadB(0);
    stashB(0);
    wait_async0();                 // barrier does not track ASYNCcnt
#else
    gloadA(0); gloadB(0);
    stashA(0); stashB(0);
#endif
    __syncthreads();

    for (int kb = 0; kb < nKB; ++kb) {
        const int cur = kb & 1;
        const bool hasNext = (kb + 1) < nKB;
        if (hasNext) {
#if ASYNC_A
            gloadA_async(kb + 1, cur ^ 1);   // DMA directly into next buffer
            gloadB(kb + 1);
#else
            gloadA(kb + 1); gloadB(kb + 1);
#endif
        }

        const __bf16* sAb = (const __bf16*)sA[cur];
        const __bf16* sBb = (const __bf16*)sB[cur];

        // A fragments: lane L -> M = l16; elems i<8: K=8*half+i ; i>=8: K=8*half+16+(i-8)
        v16bf afr[4];
        #pragma unroll
        for (int mt = 0; mt < 4; ++mt) {
            const __bf16* p = sAb + (wr * 64 + mt * 16 + l16) * APITCH + 8 * half;
            #pragma unroll
            for (int i = 0; i < 8; ++i) { afr[mt][i] = p[i]; afr[mt][i + 8] = p[i + 16]; }
        }
        // B fragments: lane L -> N = l16; elems i<8: K=8*half+i ; i>=8: K=8*half+16+(i-8)
        v16bf bfr[4];
        #pragma unroll
        for (int nt = 0; nt < 4; ++nt) {
            const __bf16* p = sBb + (wc * 64 + nt * 16 + l16) * BPITCH + 8 * half;
            #pragma unroll
            for (int i = 0; i < 8; ++i) { bfr[nt][i] = p[i]; bfr[nt][i + 8] = p[i + 16]; }
        }

        #pragma unroll
        for (int mt = 0; mt < 4; ++mt)
            #pragma unroll
            for (int nt = 0; nt < 4; ++nt)
                acc[mt][nt] = __builtin_amdgcn_wmma_f32_16x16x32_bf16(
                    false, afr[mt], false, bfr[nt], (short)0, acc[mt][nt], false, false);

        if (hasNext) {
#if ASYNC_A
            stashB(cur ^ 1);
            wait_async0();          // A DMA for next buffer must land before barrier
#else
            stashA(cur ^ 1); stashB(cur ^ 1);
#endif
        }
        __syncthreads();
    }

    // ---- epilogue ----
    // C/D layout: VGPR v, lane half h: M = v + 8h, N = l16
    #pragma unroll
    for (int mt = 0; mt < 4; ++mt) {
        const int rBase = rowBase + wr * 64 + mt * 16 + 8 * half;
        #pragma unroll
        for (int nt = 0; nt < 4; ++nt) {
            const int col = colBase + wc * 64 + nt * 16 + l16;
            #pragma unroll
            for (int v = 0; v < 8; ++v) {
                const int row = rBase + v;
                const float x = acc[mt][nt][v];
                if (EPI == 0) {
                    const float s = x / (1.0f + __expf(-x));          // SiLU
                    ((__bf16*)outPtr)[(size_t)row * outLd + col] = (__bf16)s;
                } else {
                    const float scale = rw[row * 8 + expert];
                    float* p = (float*)outPtr + (size_t)row * outLd + col;
                    const float val = scale * x;
                    if (accum) *p += val; else *p = val;
                }
            }
        }
    }
}

// x (f32) -> bf16, packed dword stores
__global__ __launch_bounds__(256) void cvt_f32_to_bf16(
    const float4* __restrict__ in, uint2* __restrict__ out, int n4)
{
    int i = blockIdx.x * blockDim.x + threadIdx.x;
    if (i < n4) {
        float4 f = in[i];
        uint2 o;
        o.x = pack2_bf16(f.x, f.y);
        o.y = pack2_bf16(f.z, f.w);
        out[i] = o;
    }
}

extern "C" void kernel_launch(void* const* d_in, const int* in_sizes, int n_in,
                              void* d_out, int out_size, void* d_ws, size_t ws_size,
                              hipStream_t stream) {
    (void)in_sizes; (void)n_in; (void)out_size; (void)ws_size;
    constexpr int E = 8, D = 1024, H = 4096, NTOK = 4096;

    const float* x  = (const float*)d_in[0];
    const float* rw = (const float*)d_in[1];
    // d_in[2] = selected_experts : unused by the reference (dense MoE)
    const float* w1 = (const float*)d_in[3];   // [E, D, H]
    const float* w2 = (const float*)d_in[4];   // [E, H, D]
    float* out = (float*)d_out;                // [NTOK, D]

    __bf16* xbf = (__bf16*)d_ws;                                              // 8 MB
    __bf16* hid = (__bf16*)((char*)d_ws + (size_t)NTOK * D * sizeof(__bf16)); // 32 MB

    // x -> bf16 once (reused by all 8 experts)
    {
        int n4 = (NTOK * D) / 4;
        cvt_f32_to_bf16<<<(n4 + 255) / 256, 256, 0, stream>>>(
            (const float4*)x, (uint2*)xbf, n4);
    }

    for (int e = 0; e < E; ++e) {
        // hidden = silu(x @ w1[e]) : M=4096, N=H=4096, K=D=1024
        moe_wmma_gemm<0><<<dim3(H / TBN, NTOK / TBM), 128, 0, stream>>>(
            xbf, w1 + (size_t)e * D * H, hid, nullptr,
            /*lda2=*/D / 2, /*ldb=*/H, /*nKB=*/D / TBK, /*outLd=*/H, e, 0);

        // out (+)= rw[:,e] * (hidden @ w2[e]) : M=4096, N=D=1024, K=H=4096
        moe_wmma_gemm<1><<<dim3(D / TBN, NTOK / TBM), 128, 0, stream>>>(
            hid, w2 + (size_t)e * H * D, out, rw,
            /*lda2=*/H / 2, /*ldb=*/D, /*nKB=*/H / TBK, /*outLd=*/D, e, /*accum=*/(e > 0));
    }
}
